// AttentionLayer_55310588838086
// MI455X (gfx1250) — matrix operand order repaired and checked by reading, not compiled, not run on hardware
//
#include <hip/hip_runtime.h>
#include <hip/hip_bf16.h>

#define Bb 4
#define Tt 4096
#define Dd 1024
#define BTD ((size_t)Bb * Tt * Dd)

typedef __attribute__((ext_vector_type(16))) __bf16 v16bf;
typedef __attribute__((ext_vector_type(8)))  float  v8f;

union FragU { v16bf v; uint4 q[2]; };
union Pack4 { __bf16 h[4]; uint2 u; };
union Pack8 { __bf16 h[8]; uint4 u; };

static __device__ __forceinline__ v8f wmma_bf16(v16bf a, v16bf b, v8f c) {
    // 8 args: (neg_a, A, neg_b, B, c_mod, C, reuse_a, reuse_b)
    return __builtin_amdgcn_wmma_f32_16x16x32_bf16(false, a, false, b,
                                                   (short)0, c, false, false);
}

static __device__ __forceinline__ v8f vzero8() {
    v8f z = {0.f, 0.f, 0.f, 0.f, 0.f, 0.f, 0.f, 0.f};
    return z;
}

// Async copy 16B global -> LDS (ASYNCcnt-tracked, no VGPR round trip).
// LDS byte address = low 32 bits of the generic shared pointer (ISA 10.2).
static __device__ __forceinline__ void async_g2l_b128(const void* gptr, void* lptr) {
    const unsigned int laddr = (unsigned int)(unsigned long long)lptr;
    asm volatile("global_load_async_to_lds_b128 %0, %1, off"
                 :: "v"(laddr), "v"(gptr) : "memory");
}
static __device__ __forceinline__ void wait_async0() {
    asm volatile("s_wait_asynccnt 0" ::: "memory");
}

// ---------------------------------------------------------------------------
// Kernel 1: QKV projection  C = x @ W^T.  z=0:Q (pre-scaled 1/sqrt(D)),
// z=1:K row-major [B,T,D], z=2:V stored transposed [B,D,T].
// 8 waves, 128x128 tile, K staged 32 at a time.  x-tile: async DMA to LDS
// (raw f32, double buffered); W-tile: reg-staged f32->bf16 (double buffered).
// One barrier per K-step; 8 WMMAs issued back-to-back per wave per step.
// ---------------------------------------------------------------------------
__global__ __launch_bounds__(256) void qkv_gemm(
    const float* __restrict__ x,
    const float* __restrict__ Wq, const float* __restrict__ Wk,
    const float* __restrict__ Wv,
    __bf16* __restrict__ Qb, __bf16* __restrict__ Kb, __bf16* __restrict__ Vt)
{
    const int z    = blockIdx.z;
    const float* W = (z == 0) ? Wq : ((z == 1) ? Wk : Wv);
    const int m0   = blockIdx.y * 128;
    const int n0   = blockIdx.x * 128;
    const int tid  = threadIdx.x;
    const int wave = tid >> 5;
    const int lane = tid & 31;
    const int lm   = lane & 15;
    const bool hif = lane >= 16;

    // x tile: raw f32, rows padded to 36 floats (144B: 16B-aligned chunks)
    __shared__ __align__(16) float  xs32[2][128][36];
    // W tile: bf16, rows padded to 40 halves (80B stride)
    __shared__ __align__(16) __bf16 wsb[2][128][40];

    const int srow = tid >> 3;        // 0..31
    const int scol = (tid & 7) * 4;   // 0,4,...,28

    v8f acc[8];
    #pragma unroll
    for (int i = 0; i < 8; ++i) acc[i] = vzero8();

    float4 wreg[4];

    // ---- prologue: stage K-step 0 into buffer 0 ---------------------------
    #pragma unroll
    for (int p = 0; p < 4; ++p) {
        const int row = p * 32 + srow;
        async_g2l_b128(x + (size_t)(m0 + row) * Dd + scol, &xs32[0][row][scol]);
        wreg[p] = *(const float4*)(W + (size_t)(n0 + row) * Dd + scol);
    }
    #pragma unroll
    for (int p = 0; p < 4; ++p) {
        const int row = p * 32 + srow;
        Pack4 pw;
        pw.h[0] = (__bf16)wreg[p].x; pw.h[1] = (__bf16)wreg[p].y;
        pw.h[2] = (__bf16)wreg[p].z; pw.h[3] = (__bf16)wreg[p].w;
        *(uint2*)&wsb[0][row][scol] = pw.u;
    }
    wait_async0();
    __syncthreads();

    for (int s = 0; s < Dd / 32; ++s) {
        const int cur  = s & 1;
        const int nxt  = cur ^ 1;
        const bool more = (s + 1) < (Dd / 32);
        const int k1   = (s + 1) * 32;

        // issue next tile's loads early (async DMA for x, vmem regs for W)
        if (more) {
            #pragma unroll
            for (int p = 0; p < 4; ++p) {
                const int row = p * 32 + srow;
                async_g2l_b128(x + (size_t)(m0 + row) * Dd + k1 + scol,
                               &xs32[nxt][row][scol]);
                wreg[p] = *(const float4*)(W + (size_t)(n0 + row) * Dd + k1 + scol);
            }
        }

        // ---- A fragment: f32 from LDS, convert to bf16 --------------------
        FragU a;
        {
            const int arow = wave * 16 + lm;
            const int kb   = hif ? 8 : 0;
            const float* base = &xs32[cur][arow][0];
            const float4 f0 = *(const float4*)(base + kb);
            const float4 f1 = *(const float4*)(base + kb + 4);
            const float4 f2 = *(const float4*)(base + kb + 16);
            const float4 f3 = *(const float4*)(base + kb + 20);
            Pack8 p0, p1;
            p0.h[0] = (__bf16)f0.x; p0.h[1] = (__bf16)f0.y;
            p0.h[2] = (__bf16)f0.z; p0.h[3] = (__bf16)f0.w;
            p0.h[4] = (__bf16)f1.x; p0.h[5] = (__bf16)f1.y;
            p0.h[6] = (__bf16)f1.z; p0.h[7] = (__bf16)f1.w;
            p1.h[0] = (__bf16)f2.x; p1.h[1] = (__bf16)f2.y;
            p1.h[2] = (__bf16)f2.z; p1.h[3] = (__bf16)f2.w;
            p1.h[4] = (__bf16)f3.x; p1.h[5] = (__bf16)f3.y;
            p1.h[6] = (__bf16)f3.z; p1.h[7] = (__bf16)f3.w;
            a.q[0] = p0.u;
            a.q[1] = p1.u;
        }

        // ---- all 8 B fragments first, then 8 WMMAs back-to-back -----------
        FragU bfr[8];
        {
            const int koff = hif ? 8 : 0;
            #pragma unroll
            for (int nt = 0; nt < 8; ++nt) {
                const int brow = nt * 16 + lm;
                bfr[nt].q[0] = *(const uint4*)&wsb[cur][brow][koff];
                bfr[nt].q[1] = *(const uint4*)&wsb[cur][brow][koff + 16];
            }
        }
        #pragma unroll
        for (int nt = 0; nt < 8; ++nt)
            acc[nt] = wmma_bf16(a.v, bfr[nt].v, acc[nt]);

        // ---- commit next W tile, drain async x copies, one barrier --------
        if (more) {
            #pragma unroll
            for (int p = 0; p < 4; ++p) {
                const int row = p * 32 + srow;
                Pack4 pw;
                pw.h[0] = (__bf16)wreg[p].x; pw.h[1] = (__bf16)wreg[p].y;
                pw.h[2] = (__bf16)wreg[p].z; pw.h[3] = (__bf16)wreg[p].w;
                *(uint2*)&wsb[nxt][row][scol] = pw.u;
            }
            wait_async0();
        }
        __syncthreads();
    }

    const float qscale = 0.03125f; // 1/sqrt(1024)
    #pragma unroll
    for (int nt = 0; nt < 8; ++nt) {
        #pragma unroll
        for (int r = 0; r < 8; ++r) {
            const int m = m0 + wave * 16 + r + (hif ? 8 : 0);
            const int n = n0 + nt * 16 + lm;
            const float v = acc[nt][r];
            if (z == 0) {
                Qb[(size_t)m * Dd + n] = (__bf16)(v * qscale);
            } else if (z == 1) {
                Kb[(size_t)m * Dd + n] = (__bf16)v;
            } else {
                const int bidx = m >> 12;     // m / 4096
                const int t    = m & 4095;
                Vt[((size_t)bidx * Dd + n) * (size_t)Tt + t] = (__bf16)v;
            }
        }
    }
}

// ---------------------------------------------------------------------------
// Kernel 2: flash attention, 16 queries per workgroup, D=1024 split across
// 8 waves (128 output cols each). Keys streamed 64 at a time from L2-resident
// K/V (16 MB per batch << 192 MB L2). Partial scores reduced through LDS;
// online softmax by wave 0; P·V accumulated per wave.
// ---------------------------------------------------------------------------
__global__ __launch_bounds__(256) void flash_attn(
    const __bf16* __restrict__ Qb, const __bf16* __restrict__ Kb,
    const __bf16* __restrict__ Vt, float* __restrict__ out)
{
    const int b    = blockIdx.y;
    const int q0   = blockIdx.x * 16;
    const int tid  = threadIdx.x;
    const int wave = tid >> 5;
    const int lane = tid & 31;
    const int lm   = lane & 15;
    const bool hif = lane >= 16;
    const int ds0  = wave * 128;   // this wave's D slice

    __shared__ __align__(16) float  sSp[8][16][64]; // per-wave partial scores
    __shared__ __align__(16) float  sS[16][64];     // reduced scores
    __shared__ __align__(16) __bf16 sP[16][64];     // softmax probs (bf16)
    __shared__ float sM[16], sL[16], sAlpha[16];

    // Q A-fragments for this wave's 128-wide D slice (scale pre-folded)
    FragU qa[4];
    {
        const __bf16* qrow = Qb + (size_t)(b * Tt + q0 + lm) * Dd + ds0;
        const int kb = hif ? 8 : 0;
        #pragma unroll
        for (int kk = 0; kk < 4; ++kk) {
            qa[kk].q[0] = *(const uint4*)(qrow + kk * 32 + kb);
            qa[kk].q[1] = *(const uint4*)(qrow + kk * 32 + kb + 16);
        }
    }

    v8f oacc[8];
    #pragma unroll
    for (int i = 0; i < 8; ++i) oacc[i] = vzero8();

    if (tid < 16) { sM[tid] = -__builtin_inff(); sL[tid] = 0.f; }
    __syncthreads();

    for (int kt = 0; kt < Tt / 64; ++kt) {
        const int key0 = kt * 64;
        if (kt + 1 < Tt / 64) {   // pull next tiles toward L2/WGP$
            __builtin_prefetch(Kb + (size_t)(b * Tt + key0 + 64 + lm) * Dd + ds0, 0, 0);
            __builtin_prefetch(Vt + (size_t)(b * Dd + ds0 + lm) * Tt + key0 + 64, 0, 0);
        }

        // ---- partial S = Q·K^T over this wave's D slice -------------------
        #pragma unroll
        for (int sub = 0; sub < 4; ++sub) {
            const __bf16* krow =
                Kb + (size_t)(b * Tt + key0 + sub * 16 + lm) * Dd + ds0;
            const int koff = hif ? 8 : 0;
            FragU kf[4];
            #pragma unroll
            for (int kk = 0; kk < 4; ++kk) {
                kf[kk].q[0] = *(const uint4*)(krow + kk * 32 + koff);
                kf[kk].q[1] = *(const uint4*)(krow + kk * 32 + koff + 16);
            }
            v8f s = vzero8();
            #pragma unroll
            for (int kk = 0; kk < 4; ++kk)
                s = wmma_bf16(qa[kk].v, kf[kk].v, s);
            #pragma unroll
            for (int r = 0; r < 8; ++r)
                sSp[wave][r + (hif ? 8 : 0)][sub * 16 + lm] = s[r];
        }
        __syncthreads();

        // ---- reduce 8 partials -> sS (256 thr x 4 floats = 1024 elems) ----
        {
            const int idx = tid * 4;
            const int row = idx >> 6;
            const int col = idx & 63;
            float4 a4 = {0.f, 0.f, 0.f, 0.f};
            #pragma unroll
            for (int w = 0; w < 8; ++w) {
                const float4 v = *(const float4*)&sSp[w][row][col];
                a4.x += v.x; a4.y += v.y; a4.z += v.z; a4.w += v.w;
            }
            *(float4*)&sS[row][col] = a4;
        }
        __syncthreads();

        // ---- online softmax update (wave 0 only) --------------------------
        if (tid < 32) {
            const int row = lm;
            const int cb  = hif ? 32 : 0;
            float sv[32];
            float mymax = -__builtin_inff();
            #pragma unroll
            for (int j = 0; j < 32; ++j) {
                sv[j] = sS[row][cb + j];
                mymax = fmaxf(mymax, sv[j]);
            }
            const float rowmax = fmaxf(mymax, __shfl_xor(mymax, 16, 32));
            const float mo = sM[row];
            const float mn = fmaxf(mo, rowmax);
            const float alpha = __expf(mo - mn);
            float sum = 0.f;
            #pragma unroll
            for (int j = 0; j < 32; ++j) {
                const float p = __expf(sv[j] - mn);
                sum += p;
                sP[row][cb + j] = (__bf16)p;
            }
            const float tot = sum + __shfl_xor(sum, 16, 32);
            if (!hif) {
                sM[row]     = mn;
                sL[row]     = sL[row] * alpha + tot;
                sAlpha[row] = alpha;
            }
        }
        __syncthreads();

        // ---- rescale O and accumulate P·V over this wave's columns --------
        float arsc[8];
        #pragma unroll
        for (int r = 0; r < 8; ++r) arsc[r] = sAlpha[r + (hif ? 8 : 0)];

        FragU pa[2];
        {
            const int kb = hif ? 8 : 0;
            #pragma unroll
            for (int h = 0; h < 2; ++h) {
                pa[h].q[0] = *(const uint4*)&sP[lm][h * 32 + kb];
                pa[h].q[1] = *(const uint4*)&sP[lm][h * 32 + kb + 16];
            }
        }
        #pragma unroll
        for (int ct = 0; ct < 8; ++ct) {
            const __bf16* vrow =
                Vt + (size_t)(b * Dd + ds0 + ct * 16 + lm) * Tt + key0;
            const int koff = hif ? 8 : 0;
            FragU bv[2];
            #pragma unroll
            for (int h = 0; h < 2; ++h) {
                bv[h].q[0] = *(const uint4*)(vrow + h * 32 + koff);
                bv[h].q[1] = *(const uint4*)(vrow + h * 32 + koff + 16);
            }
            #pragma unroll
            for (int r = 0; r < 8; ++r) oacc[ct][r] *= arsc[r];
            #pragma unroll
            for (int h = 0; h < 2; ++h)
                oacc[ct] = wmma_bf16(pa[h].v, bv[h].v, oacc[ct]);
        }
    }
    __syncthreads();

    // ---- normalize by running denominator and store -----------------------
    float linv[8];
    #pragma unroll
    for (int r = 0; r < 8; ++r) linv[r] = 1.0f / sL[r + (hif ? 8 : 0)];
    #pragma unroll
    for (int ct = 0; ct < 8; ++ct) {
        #pragma unroll
        for (int r = 0; r < 8; ++r) {
            const int row = q0 + r + (hif ? 8 : 0);
            const int col = ds0 + ct * 16 + lm;
            out[(size_t)(b * Tt + row) * Dd + col] = oacc[ct][r] * linv[r];
        }
    }
}

extern "C" void kernel_launch(void* const* d_in, const int* in_sizes, int n_in,
                              void* d_out, int out_size, void* d_ws, size_t ws_size,
                              hipStream_t stream) {
    (void)in_sizes; (void)n_in; (void)out_size; (void)ws_size;
    const float* x  = (const float*)d_in[0];
    const float* Wq = (const float*)d_in[1];
    const float* Wk = (const float*)d_in[2];
    const float* Wv = (const float*)d_in[3];

    __bf16* Qb = (__bf16*)d_ws;         // [B,T,D] bf16, pre-scaled
    __bf16* Kb = Qb + BTD;              // [B,T,D] bf16
    __bf16* Vt = Kb + BTD;              // [B,D,T] bf16 (transposed)
    float*  out = (float*)d_out;

    dim3 g1(Dd / 128, (Bb * Tt) / 128, 3);
    qkv_gemm<<<g1, 256, 0, stream>>>(x, Wq, Wk, Wv, Qb, Kb, Vt);

    dim3 g2(Tt / 16, Bb);
    flash_attn<<<g2, 256, 0, stream>>>(Qb, Kb, Vt, out);
}
